// MultiHeadSelfAttention_22127671509168
// MI455X (gfx1250) — matrix operand order repaired and checked by reading, not compiled, not run on hardware
//
#include <hip/hip_runtime.h>
#include <math.h>

// ---------------------------------------------------------------------------
// CDNA5 (gfx1250, wave32) WMMA helpers: V_WMMA_F32_16X16X32_F16
// ---------------------------------------------------------------------------
typedef __attribute__((ext_vector_type(16))) _Float16 v16h;
typedef __attribute__((ext_vector_type(8)))  float    v8f;
typedef __attribute__((ext_vector_type(4)))  int      v4i;

__device__ __forceinline__ v8f wmma_f32_f16(v16h a, v16h b, v8f c) {
  // (neg_a, A, neg_b, B, c_mod, C, reuse_a, reuse_b)
  return __builtin_amdgcn_wmma_f32_16x16x32_f16(false, a, false, b, (short)0, c,
                                                false, false);
}

// A fragment: 16x32 (MxK), row-major source, leading dim lda.
// Lane&15 = row; lane half selects K {0..7,16..23} vs {8..15,24..31}.
// Per lane: two contiguous 16-byte runs -> 2x b128 loads.
__device__ __forceinline__ v16h load_a_frag(const _Float16* __restrict__ A,
                                            int lda, int lane) {
  const int row = lane & 15, hi = lane >> 4;
  const _Float16* p = A + (long)row * lda;
  v16h a;
#pragma unroll
  for (int v = 0; v < 8; ++v) {
    const int kb = 2 * v + ((v >= 4) ? 8 : 0) + hi * 8;
    a[2 * v]     = p[kb];
    a[2 * v + 1] = p[kb + 1];
  }
  return a;
}

// B fragment: 32x16 (KxN) from K-contiguous storage: element(k,col) =
// Bt[col*ldk + k]. Lane&15 = col; lane half selects K {0..7,16..23} vs
// {8..15,24..31} -> 2x b128 loads (global or ds).
__device__ __forceinline__ v16h load_b_fragT(const _Float16* Bt, long ldk,
                                             int lane) {
  const int col = lane & 15, hi = lane >> 4;
  const _Float16* p = Bt + (long)col * ldk;
  v16h b;
#pragma unroll
  for (int i = 0; i < 16; ++i) b[i] = p[(i & 7) + hi * 8 + ((i >> 3) << 4)];
  return b;
}

// ---------------------------------------------------------------------------
// Async global->LDS staging (CDNA5 ASYNCcnt path), with sync fallback.
// ROCm builtin signature takes v4i pointers: AS(1) src, AS(3) dst.
// ---------------------------------------------------------------------------
#if defined(__gfx1250__) && \
    __has_builtin(__builtin_amdgcn_global_load_async_to_lds_b128)
#define FA_HAS_ASYNC 1
typedef __attribute__((address_space(1))) v4i as1_v4i;
typedef __attribute__((address_space(3))) v4i as3_v4i;
#else
#define FA_HAS_ASYNC 0
#endif

__device__ __forceinline__ void stage16(const _Float16* gsrc,
                                        _Float16* ldst) {
#if FA_HAS_ASYNC
  __builtin_amdgcn_global_load_async_to_lds_b128((as1_v4i*)gsrc,
                                                 (as3_v4i*)ldst, 0, 0);
#else
  *(float4*)ldst = *(const float4*)gsrc;
#endif
}

__device__ __forceinline__ void stage_wait() {
#if FA_HAS_ASYNC
  asm volatile("s_wait_asynccnt 0" ::: "memory");
#endif
}

// ---------------------------------------------------------------------------
// Conversion kernels
// ---------------------------------------------------------------------------
__global__ void cvt_f32_to_f16(const float* __restrict__ in,
                               _Float16* __restrict__ out, long n) {
  long i = (long)blockIdx.x * 256 + threadIdx.x;
  if (i < n) out[i] = (_Float16)in[i];
}

// in [K,N] row-major fp32 -> out [N,K] row-major f16
__global__ void cvt_transpose_f16(const float* __restrict__ in,
                                  _Float16* __restrict__ out, int K, int N) {
  long i = (long)blockIdx.x * 256 + threadIdx.x;
  if (i < (long)K * N) {
    const int k = (int)(i / N);
    const int n = (int)(i % N);
    out[(long)n * K + k] = (_Float16)in[i];
  }
}

// ---------------------------------------------------------------------------
// GEMM: Y[M,N] = A[M,K] @ Bt[N,K]^T + bias, f16 in, f32 accum.
// Each wave owns a 64x64 tile: 16 accumulators, 16 WMMAs per 32-deep k-step.
// MODE 0: f16 row-major.  MODE 1: f32 row-major.  MODE 2: f16 head-transposed
// (out[(b*N + c)*S + s], row = b*S + s, S = 1<<slog2) -> emits V^T.
// ---------------------------------------------------------------------------
#define GEMM_WAVES 8

template <int MODE>
__global__ __launch_bounds__(256) void gemm_wmma(
    const _Float16* __restrict__ A, const _Float16* __restrict__ Bt,
    const float* __restrict__ bias, void* __restrict__ outp, int M, int N,
    int K, int slog2) {
  const int lane = threadIdx.x & 31;
  const int wave = threadIdx.x >> 5;
  const int col = lane & 15, hi = lane >> 4;
  const int m0 = (blockIdx.x * GEMM_WAVES + wave) * 64;
  const int n0 = blockIdx.y * 64;
  if (m0 >= M) return;

  v8f acc[4][4] = {};  // [m subtile][n subtile]

  const _Float16* arow0 = A + (long)(m0 + 0 * 16) * K;
  const _Float16* arow1 = A + (long)(m0 + 1 * 16) * K;
  const _Float16* arow2 = A + (long)(m0 + 2 * 16) * K;
  const _Float16* arow3 = A + (long)(m0 + 3 * 16) * K;

  for (int k0 = 0; k0 < K; k0 += 32) {
    if (k0 + 64 < K) __builtin_prefetch(arow0 + k0 + 64, 0, 1);
    v16h a[4];
    a[0] = load_a_frag(arow0 + k0, K, lane);
    a[1] = load_a_frag(arow1 + k0, K, lane);
    a[2] = load_a_frag(arow2 + k0, K, lane);
    a[3] = load_a_frag(arow3 + k0, K, lane);
#pragma unroll
    for (int t = 0; t < 4; ++t) {
      v16h b = load_b_fragT(Bt + (long)(n0 + t * 16) * K + k0, K, lane);
#pragma unroll
      for (int mi = 0; mi < 4; ++mi)
        acc[mi][t] = wmma_f32_f16(a[mi], b, acc[mi][t]);
    }
  }

  // Epilogue. C layout: VGPR r holds (row = r + 8*hi, col = lane&15).
#pragma unroll
  for (int mi = 0; mi < 4; ++mi) {
#pragma unroll
    for (int t = 0; t < 4; ++t) {
#pragma unroll
      for (int r = 0; r < 8; ++r) {
        const int row = m0 + mi * 16 + r + hi * 8;
        const int c = n0 + t * 16 + col;
        const float v = acc[mi][t][r] + bias[c];
        if (MODE == 0) {
          ((_Float16*)outp)[(long)row * N + c] = (_Float16)v;
        } else if (MODE == 1) {
          ((float*)outp)[(long)row * N + c] = v;
        } else {
          const int bb = row >> slog2;
          const int s = row - (bb << slog2);
          ((_Float16*)outp)[(((long)bb * N + c) << slog2) + s] = (_Float16)v;
        }
      }
    }
  }
}

// ---------------------------------------------------------------------------
// Flash attention.
// Block = 8 waves = 128 queries of one (batch, head). Per 64-key step the
// block stages K (64x64, [key][d]) and V^T (64x64, [d][key]) into LDS with
// async global->LDS copies, then every wave runs 16 WMMAs (8 scores + 8 PV)
// from LDS. Causal loop bound is block-uniform; per-wave compute is guarded.
// ---------------------------------------------------------------------------
#define FA_WAVES 8
#define KPAD 72  // 64 + 8 halfs row padding (16B-aligned, dodges bank conflicts)

__global__ __launch_bounds__(256) void flash_attn(
    const _Float16* __restrict__ Qh, const _Float16* __restrict__ Kh,
    const _Float16* __restrict__ Vt, _Float16* __restrict__ atth, int S,
    int Hdim, int hd) {
  __shared__ _Float16 lds_k[64][KPAD];           // [key][d]    9 KB
  __shared__ _Float16 lds_v[64][KPAD];           // [d][key]    9 KB
  __shared__ _Float16 lds_p[FA_WAVES][16][64];   // P tiles    16 KB

  const int tid = threadIdx.x;
  const int lane = tid & 31;
  const int wave = tid >> 5;
  const int col = lane & 15, hi = lane >> 4;
  const int qb = blockIdx.x * (FA_WAVES * 16);
  const int q0 = qb + wave * 16;
  const int h = blockIdx.y;
  const int b = blockIdx.z;
  const long base = (long)b * S * Hdim;
  const int hoff = h * hd;
  const _Float16* VtH = Vt + (long)(b * Hdim + hoff) * S;
  const float rscale = 1.0f / sqrtf((float)hd);

  // Q fragments: 16x64 = two 16x32 k-chunks, loaded once from global.
  const _Float16* qp = Qh + base + (long)q0 * Hdim + hoff;
  const v16h aq0 = load_a_frag(qp, Hdim, lane);
  const v16h aq1 = load_a_frag(qp + 32, Hdim, lane);

  v8f acc[4] = {};
  float mrow[8], lrow[8];
#pragma unroll
  for (int r = 0; r < 8; ++r) { mrow[r] = -1e30f; lrow[r] = 0.0f; }

  const int qmaxk = qb + FA_WAVES * 16 - 1;  // block-uniform causal bound
  for (int key0 = 0; key0 <= qmaxk; key0 += 64) {
    // ---- cooperative async staging: 512 x 16B chunks per tile, 2/thread
#pragma unroll
    for (int i = 0; i < 2; ++i) {
      const int c = tid + i * 256;
      const int row = c >> 3;            // 0..63
      const int off = (c & 7) * 8;       // halfs: 0,8,...,56
      stage16(Kh + base + (long)(key0 + row) * Hdim + hoff + off,
              &lds_k[row][off]);
      stage16(VtH + (long)row * S + key0 + off, &lds_v[row][off]);
    }
    stage_wait();
    __syncthreads();

    if (key0 <= q0 + 15) {  // wave-uniform guard (no barriers inside)
      // ---- scores: four 16x16 tiles from LDS K (2 chained WMMAs each)
      v8f sc[4];
#pragma unroll
      for (int j = 0; j < 4; ++j) {
        v16h bk0 = load_b_fragT(&lds_k[j * 16][0], KPAD, lane);
        v16h bk1 = load_b_fragT(&lds_k[j * 16][32], KPAD, lane);
        v8f z = {};
        z = wmma_f32_f16(aq0, bk0, z);
        sc[j] = wmma_f32_f16(aq1, bk1, z);
      }

      // ---- online softmax in C layout (row r+8*hi in one 16-lane half)
#pragma unroll
      for (int r = 0; r < 8; ++r) {
        const int qg = q0 + r + hi * 8;
        float s[4];
        float mx = -1e30f;
#pragma unroll
        for (int j = 0; j < 4; ++j) {
          const int kg = key0 + j * 16 + col;
          s[j] = (kg <= qg) ? sc[j][r] * rscale : -1e30f;
          mx = fmaxf(mx, s[j]);
        }
#pragma unroll
        for (int off = 8; off; off >>= 1)
          mx = fmaxf(mx, __shfl_xor(mx, off, 32));

        const float mnew = fmaxf(mrow[r], mx);
        const float alpha = __expf(mrow[r] - mnew);
        mrow[r] = mnew;

        float rs = 0.0f;
        float p[4];
#pragma unroll
        for (int j = 0; j < 4; ++j) {
          p[j] = __expf(s[j] - mnew);
          rs += p[j];
        }
#pragma unroll
        for (int off = 8; off; off >>= 1) rs += __shfl_xor(rs, off, 32);
        lrow[r] = lrow[r] * alpha + rs;

#pragma unroll
        for (int t = 0; t < 4; ++t) acc[t][r] *= alpha;

#pragma unroll
        for (int j = 0; j < 4; ++j)
          lds_p[wave][r + hi * 8][j * 16 + col] = (_Float16)p[j];
      }

      // ---- reload P as two A fragments (wave-private LDS; DS in-order)
      v16h pa[2];
      {
        const int arow = lane & 15;
#pragma unroll
        for (int c2 = 0; c2 < 2; ++c2) {
#pragma unroll
          for (int v = 0; v < 8; ++v) {
            const int kb = c2 * 32 + 2 * v + ((v >= 4) ? 8 : 0) + hi * 8;
            pa[c2][2 * v]     = lds_p[wave][arow][kb];
            pa[c2][2 * v + 1] = lds_p[wave][arow][kb + 1];
          }
        }
      }

      // ---- P @ V from LDS V^T (keys contiguous)
#pragma unroll
      for (int t = 0; t < 4; ++t) {
#pragma unroll
        for (int c2 = 0; c2 < 2; ++c2) {
          v16h bv = load_b_fragT(&lds_v[t * 16][c2 * 32], KPAD, lane);
          acc[t] = wmma_f32_f16(pa[c2], bv, acc[t]);
        }
      }
    }
    __syncthreads();  // protect LDS tiles before next staging pass
  }

  // ---- normalize and store att (f16, feeds the mix GEMM)
#pragma unroll
  for (int r = 0; r < 8; ++r) {
    const float inv = 1.0f / lrow[r];
    const int row = q0 + r + hi * 8;
#pragma unroll
    for (int t = 0; t < 4; ++t) {
      atth[base + (long)row * Hdim + hoff + t * 16 + col] =
          (_Float16)(acc[t][r] * inv);
    }
  }
}

// ---------------------------------------------------------------------------
// Launch
// ---------------------------------------------------------------------------
extern "C" void kernel_launch(void* const* d_in, const int* in_sizes, int n_in,
                              void* d_out, int out_size, void* d_ws,
                              size_t ws_size, hipStream_t stream) {
  (void)in_sizes; (void)n_in; (void)out_size; (void)ws_size;
  const float* x    = (const float*)d_in[0];
  const float* Wq   = (const float*)d_in[1];
  const float* bq   = (const float*)d_in[2];
  const float* Wk   = (const float*)d_in[3];
  const float* bk   = (const float*)d_in[4];
  const float* Wv   = (const float*)d_in[5];
  const float* bv   = (const float*)d_in[6];
  const float* Wmix = (const float*)d_in[7];
  const float* bmix = (const float*)d_in[8];

  const int B = 4, S = 2048, D = 1024, H = 1024, nh = 16, hd = 64;
  const int SLOG2 = 11;        // S == 2048
  const long M = (long)B * S;  // 8192

  // Workspace (f16): x 16MB | 4 transposed weights 2MB each | Q 16MB |
  // K 16MB | Vt 16MB | att 16MB  => 88 MB total.
  char* ws = (char*)d_ws;
  _Float16* xh    = (_Float16*)(ws);
  _Float16* wqT   = (_Float16*)(ws + (16L << 20));
  _Float16* wkT   = (_Float16*)(ws + (18L << 20));
  _Float16* wvT   = (_Float16*)(ws + (20L << 20));
  _Float16* wmixT = (_Float16*)(ws + (22L << 20));
  _Float16* Qh    = (_Float16*)(ws + (24L << 20));
  _Float16* Kh    = (_Float16*)(ws + (40L << 20));
  _Float16* Vth   = (_Float16*)(ws + (56L << 20));
  _Float16* atth  = (_Float16*)(ws + (72L << 20));

  const long nx = M * D;
  cvt_f32_to_f16<<<(unsigned)((nx + 255) / 256), 256, 0, stream>>>(x, xh, nx);
  const long nw = (long)D * H;
  const unsigned wblk = (unsigned)((nw + 255) / 256);
  cvt_transpose_f16<<<wblk, 256, 0, stream>>>(Wq, wqT, D, H);
  cvt_transpose_f16<<<wblk, 256, 0, stream>>>(Wk, wkT, D, H);
  cvt_transpose_f16<<<wblk, 256, 0, stream>>>(Wv, wvT, D, H);
  cvt_transpose_f16<<<wblk, 256, 0, stream>>>(Wmix, wmixT, D, H);

  // 64x64 tile per wave, 8 waves per block: grid (8192/512, 1024/64).
  dim3 gg((unsigned)(M / (64 * GEMM_WAVES)), (unsigned)(H / 64));  // (16,16)
  gemm_wmma<0><<<gg, 256, 0, stream>>>(xh, wqT, bq, Qh, (int)M, H, D, SLOG2);
  gemm_wmma<0><<<gg, 256, 0, stream>>>(xh, wkT, bk, Kh, (int)M, H, D, SLOG2);
  gemm_wmma<2><<<gg, 256, 0, stream>>>(xh, wvT, bv, Vth, (int)M, H, D, SLOG2);

  dim3 fg((unsigned)(S / (16 * FA_WAVES)), (unsigned)nh, (unsigned)B);
  flash_attn<<<fg, 256, 0, stream>>>(Qh, Kh, Vth, atth, S, H, hd);

  gemm_wmma<1><<<gg, 256, 0, stream>>>(atth, wmixT, bmix, d_out, (int)M, H, D,
                                       SLOG2);
}
